// Manhattan_Attention_FUsion_34497177322055
// MI455X (gfx1250) — matrix operand order repaired and checked by reading, not compiled, not run on hardware
//
#include <hip/hip_runtime.h>
#include <hip/hip_fp16.h>
#include <hip/hip_bf16.h>

// ---------------------------------------------------------------------------
// Manhattan attention fusion for MI455X (gfx1250, wave32, WMMA).
// Pipeline: convert_w -> qkv_gemm (fused bias + rotary theta-shift epilogue)
//           -> attn_pass (W axis) -> attn_pass (H axis) -> lepe depthwise conv
//           -> out_gemm (bias + NCHW store).
// All matmuls use v_wmma_f32_16x16x32_f16 (f16 operands, f32 accumulate).
// ---------------------------------------------------------------------------

typedef __attribute__((ext_vector_type(16))) _Float16 v16h;
typedef __attribute__((ext_vector_type(8)))  _Float16 v8h;
typedef __attribute__((ext_vector_type(8)))  float    v8f;

#define NB   16
#define NC   256
#define NHW  4096          // 64*64 spatial per image
#define ROWS 65536         // 16 * 4096
#define NHEAD 4
#define DK   64

__device__ __forceinline__ v8f wmma_f16(v16h a, v16h b, v8f c) {
  return __builtin_amdgcn_wmma_f32_16x16x32_f16(
      /*neg_a=*/false, a, /*neg_b=*/false, b,
      /*c_mod=*/(short)0, c, /*reuse_a=*/false, /*reuse_b=*/false);
}

__device__ __forceinline__ v8f v8f_zero() {
  v8f z;
#pragma unroll
  for (int i = 0; i < 8; ++i) z[i] = 0.0f;
  return z;
}

// A-matrix fragment (16 x 32, f16), row-major source with given stride (halfs).
// Per ISA 7.12.2: lanes 0-15 hold K={0..7,16..23}, lanes 16-31 K={8..15,24..31}.
__device__ __forceinline__ v16h load_a_frag(const _Float16* buf, int row0,
                                            int k0, int stride) {
  const int lane = threadIdx.x & 31;
  const _Float16* p =
      buf + (row0 + (lane & 15)) * stride + k0 + ((lane & 16) ? 8 : 0);
  v8h lo = *(const v8h*)p;
  v8h hi = *(const v8h*)(p + 16);
  return __builtin_shufflevector(lo, hi, 0, 1, 2, 3, 4, 5, 6, 7, 8, 9, 10, 11,
                                 12, 13, 14, 15);
}

// B-matrix fragment (32 x 16, f16) for D = A*B with B = R^T, R row-major
// (lane holds one column n => one row of R). Same K order as A:
// lanes 0-15 hold K={0..7,16..23}, lanes 16-31 K={8..15,24..31}.
__device__ __forceinline__ v16h load_b_frag(const _Float16* buf, int n0,
                                            int k0, int stride) {
  const int lane = threadIdx.x & 31;
  const _Float16* p =
      buf + (n0 + (lane & 15)) * stride + k0 + ((lane & 16) ? 8 : 0);
  v8h lo = *(const v8h*)p;
  v8h hi = *(const v8h*)(p + 16);
  return __builtin_shufflevector(lo, hi, 0, 1, 2, 3, 4, 5, 6, 7, 8, 9, 10, 11,
                                 12, 13, 14, 15);
}

// ---------------------------------------------------------------------------
// Weight conversion: f32 -> f16, fold dk^-0.5 into wk.
// ---------------------------------------------------------------------------
__global__ void convert_w(const float* __restrict__ wq,
                          const float* __restrict__ wk,
                          const float* __restrict__ wv,
                          const float* __restrict__ wo,
                          _Float16* __restrict__ wq16,
                          _Float16* __restrict__ wk16,
                          _Float16* __restrict__ wv16,
                          _Float16* __restrict__ wo16) {
  int i = blockIdx.x * 256 + threadIdx.x;
  if (i < NC * NC) {
    wq16[i] = (_Float16)wq[i];
    wk16[i] = (_Float16)(wk[i] * 0.125f);  // scaling = dk^-0.5 = 0.125
    wv16[i] = (_Float16)wv[i];
    wo16[i] = (_Float16)wo[i];
  }
}

// ---------------------------------------------------------------------------
// QKV GEMM.  grid = (1024, 4, 3); block = 128 (4 waves).
// blockIdx.z: 0 -> q (from x1), 1 -> k (from x2, prescaled), 2 -> v (from x2)
// Each block computes a 64x64 tile of (X_nhwc @ W).  A is gathered from NCHW
// f32 and converted to f16 in LDS.  Epilogue: bias, rotary theta-shift
// (q,k via __shfl_xor lane pairing) and f16 stores in attention layouts.
// ---------------------------------------------------------------------------
__global__ __launch_bounds__(128) void qkv_gemm(
    const float* __restrict__ x1, const float* __restrict__ x2,
    const _Float16* __restrict__ wq16, const _Float16* __restrict__ wk16,
    const _Float16* __restrict__ wv16, const float* __restrict__ bq,
    const float* __restrict__ bk, const float* __restrict__ bv,
    _Float16* __restrict__ qr16, _Float16* __restrict__ kr16,
    _Float16* __restrict__ v16) {
  const int matsel = blockIdx.z;
  const float* X = (matsel == 0) ? x1 : x2;
  const _Float16* Wm = (matsel == 0) ? wq16 : ((matsel == 1) ? wk16 : wv16);
  const float* bias = (matsel == 0) ? bq : ((matsel == 1) ? bk : bv);

  const int row0 = blockIdx.x * 64;  // global row (b*4096 + hw)
  const int n0 = blockIdx.y * 64;    // output channel base
  const int img = row0 >> 12;
  const int hw0 = row0 & 4095;

  __shared__ _Float16 As[64 * 40];  // [m][kk], stride 40 halfs (16B aligned)
  __shared__ _Float16 Bs[64 * 40];  // [n][kk]

  const int t = threadIdx.x;
  const int wave = t >> 5;
  const int lane = t & 31;

  v8f acc[4];
#pragma unroll
  for (int j = 0; j < 4; ++j) acc[j] = v8f_zero();

  for (int k0 = 0; k0 < NC; k0 += 32) {
    if (k0 + 32 < NC)  // prefetch next A panel (global_prefetch_b8 path)
      __builtin_prefetch(&X[((size_t)img * NC + (k0 + 32)) * NHW + hw0], 0, 3);
    // A tile: 64 rows x 32 k from X[img][k][hw]; coalesced over m (=hw).
    for (int e = t; e < 64 * 32; e += 128) {
      int kk = e >> 6, m = e & 63;
      float xv = X[((size_t)img * NC + (k0 + kk)) * NHW + hw0 + m];
      As[m * 40 + kk] = (_Float16)xv;
    }
    // W tile transposed: Bs[n][kk] = W[k0+kk][n0+n]; coalesced over n.
    for (int e = t; e < 64 * 32; e += 128) {
      int kk = e >> 6, nn = e & 63;
      Bs[nn * 40 + kk] = Wm[(size_t)(k0 + kk) * NC + n0 + nn];
    }
    __syncthreads();
    v16h a = load_a_frag(As, wave * 16, 0, 40);
#pragma unroll
    for (int j = 0; j < 4; ++j) {
      v16h b = load_b_frag(Bs, j * 16, 0, 40);
      acc[j] = wmma_f16(a, b, acc[j]);
    }
    __syncthreads();
  }

  // Epilogue.  C fragment: col = n0+j*16+(lane&15); rows = (lane>>4)*8 + r.
  const int colL = lane & 15;
  const int rbase = (lane >> 4) * 8;
  const int mrow0 = row0 + wave * 16;
#pragma unroll
  for (int j = 0; j < 4; ++j) {
    const int c = n0 + j * 16 + colL;
    float bval = bias[c];
    if (matsel == 1) bval *= 0.125f;  // k bias also scaled
    if (matsel == 2) {
#pragma unroll
      for (int r = 0; r < 8; ++r) {
        int row = mrow0 + rbase + r;
        v16[(size_t)row * NC + c] = (_Float16)(acc[j][r] + bval);
      }
    } else {
      // rotary theta-shift: pairs (c even, c odd) live in lanes l, l^1.
      const int nh = c >> 6;
      const int d = c & 63;
      const float ang =
          __powf(10000.0f, -(float)(d >> 1) * (1.0f / 31.0f));
      _Float16* dst = (matsel == 0) ? qr16 : kr16;
#pragma unroll
      for (int r = 0; r < 8; ++r) {
        int row = mrow0 + rbase + r;
        int hw = row & 4095;
        int bimg = row >> 12;
        float val = acc[j][r] + bval;
        float partner = __shfl_xor(val, 1, 32);
        float rot = (c & 1) ? partner : -partner;
        float s, co;
        __sincosf((float)hw * ang, &s, &co);
        float res = val * co + rot * s;
        dst[(((size_t)bimg * NHEAD + nh) * NHW + hw) * DK + d] = (_Float16)res;
      }
    }
  }
}

// ---------------------------------------------------------------------------
// Axial attention pass, 4096 blocks of 128 threads (one 64x64x64 attention
// problem per block).  pass 0 = along W (per (b,h,n)); pass 1 = along H
// (per (b,w,n)).  qr/kr layout: [b][n][hw][d].  v2 layout: [b][n][x][h][d].
// attn out layout: NHWC f16 [b][hw][c].
// ---------------------------------------------------------------------------
__global__ __launch_bounds__(128) void attn_pass(
    const _Float16* __restrict__ qr16, const _Float16* __restrict__ kr16,
    const _Float16* __restrict__ vsrc, _Float16* __restrict__ outp,
    const int pass) {
  const int bid = blockIdx.x;
  const int nh = bid & 3;
  const int sl = (bid >> 2) & 63;  // h (pass 0) or w (pass 1)
  const int b = bid >> 8;

  size_t qkBase, vBase, oBase;
  int qkStride, vStride, oStride;
  if (pass == 0) {
    qkBase = (((size_t)b * NHEAD + nh) * NHW + sl * 64) * DK;  qkStride = 64;
    vBase  = ((size_t)b * NHW + sl * 64) * NC + nh * DK;       vStride = 256;
    oBase  = (((size_t)b * NHEAD + nh) * NHW + sl) * DK;       oStride = 4096;
  } else {
    qkBase = (((size_t)b * NHEAD + nh) * NHW + sl) * DK;       qkStride = 4096;
    vBase  = (((size_t)b * NHEAD + nh) * 64 + sl) * 4096;      vStride = 64;
    oBase  = ((size_t)b * NHW + sl) * NC + nh * DK;            oStride = 16384;
  }

  __shared__ _Float16 Qs[64 * 72];
  __shared__ _Float16 Ks[64 * 72];
  __shared__ _Float16 VTs[64 * 72];  // transposed: VT[d][y]
  __shared__ float    Ss[64 * 66];
  __shared__ _Float16 Ps[64 * 72];

  const int t = threadIdx.x;
  const int wave = t >> 5;
  const int lane = t & 31;

  for (int e = t; e < 64 * 64; e += 128) {
    int r = e >> 6, d = e & 63;
    Qs[r * 72 + d] = qr16[qkBase + (size_t)r * qkStride + d];
    Ks[r * 72 + d] = kr16[qkBase + (size_t)r * qkStride + d];
    VTs[d * 72 + r] = vsrc[vBase + (size_t)r * vStride + d];
  }
  __syncthreads();

  const int m0 = wave * 16;
  // decay = log(1 - 2^(-1 - 3*nh/4));  2^x = e^(x*ln2)
  const float decay =
      __logf(1.0f - __expf((-1.0f - 3.0f * (float)nh * 0.25f) * 0.6931471805599453f));

  // S = Qr @ Kr^T + |x-y| * decay
  {
    v16h a0 = load_a_frag(Qs, m0, 0, 72);
    v16h a1 = load_a_frag(Qs, m0, 32, 72);
#pragma unroll
    for (int j = 0; j < 4; ++j) {
      v8f acc = v8f_zero();
      v16h b0 = load_b_frag(Ks, j * 16, 0, 72);
      v16h b1 = load_b_frag(Ks, j * 16, 32, 72);
      acc = wmma_f16(a0, b0, acc);
      acc = wmma_f16(a1, b1, acc);
      const int y = j * 16 + (lane & 15);
      const int xb = m0 + ((lane >> 4) * 8);
#pragma unroll
      for (int r = 0; r < 8; ++r) {
        int x = xb + r;
        Ss[x * 66 + y] = acc[r] + decay * fabsf((float)(x - y));
      }
    }
  }
  __syncthreads();

  // row softmax -> P (f16)
  if (t < 64) {
    float m = -1e30f;
    for (int y = 0; y < 64; ++y) m = fmaxf(m, Ss[t * 66 + y]);
    float sum = 0.0f;
    for (int y = 0; y < 64; ++y) sum += __expf(Ss[t * 66 + y] - m);
    float inv = 1.0f / sum;
    for (int y = 0; y < 64; ++y)
      Ps[t * 72 + y] = (_Float16)(__expf(Ss[t * 66 + y] - m) * inv);
  }
  __syncthreads();

  // O = P @ V   (B fragment from VT so columns of V are contiguous)
  {
    v16h a0 = load_a_frag(Ps, m0, 0, 72);
    v16h a1 = load_a_frag(Ps, m0, 32, 72);
#pragma unroll
    for (int j = 0; j < 4; ++j) {
      v8f acc = v8f_zero();
      v16h b0 = load_b_frag(VTs, j * 16, 0, 72);
      v16h b1 = load_b_frag(VTs, j * 16, 32, 72);
      acc = wmma_f16(a0, b0, acc);
      acc = wmma_f16(a1, b1, acc);
      const int d = j * 16 + (lane & 15);
      const int xb = m0 + ((lane >> 4) * 8);
#pragma unroll
      for (int r = 0; r < 8; ++r) {
        int x = xb + r;
        outp[oBase + (size_t)x * oStride + d] = (_Float16)acc[r];
      }
    }
  }
}

// ---------------------------------------------------------------------------
// Depthwise 5x5 conv (lepe) on v, accumulated into attention output (NHWC f16)
// ---------------------------------------------------------------------------
__global__ void lepe_add(const _Float16* __restrict__ v16,
                         const float* __restrict__ w_dw,
                         const float* __restrict__ b_dw,
                         _Float16* __restrict__ attn16) {
  int idx = blockIdx.x * 256 + threadIdx.x;  // [b][h][w][c]
  int c = idx & 255;
  int w = (idx >> 8) & 63;
  int h = (idx >> 14) & 63;
  int b = idx >> 20;
  float acc = b_dw[c];
#pragma unroll
  for (int i = 0; i < 5; ++i) {
    int hh = h + i - 2;
    if (hh < 0 || hh > 63) continue;
#pragma unroll
    for (int j = 0; j < 5; ++j) {
      int ww = w + j - 2;
      if (ww < 0 || ww > 63) continue;
      acc += (float)v16[((size_t)b * NHW + hh * 64 + ww) * NC + c] *
             w_dw[(i * 5 + j) * NC + c];
    }
  }
  attn16[idx] = (_Float16)((float)attn16[idx] + acc);
}

// ---------------------------------------------------------------------------
// Output projection: (attn + lepe) @ wo + bo, stored NCHW f32.
// grid = (1024, 4), block = 128.
// ---------------------------------------------------------------------------
__global__ __launch_bounds__(128) void out_gemm(
    const _Float16* __restrict__ a16, const _Float16* __restrict__ wo16,
    const float* __restrict__ bo, float* __restrict__ out) {
  const int row0 = blockIdx.x * 64;
  const int n0 = blockIdx.y * 64;

  __shared__ _Float16 As[64 * 40];
  __shared__ _Float16 Bs[64 * 40];

  const int t = threadIdx.x;
  const int wave = t >> 5;
  const int lane = t & 31;

  v8f acc[4];
#pragma unroll
  for (int j = 0; j < 4; ++j) acc[j] = v8f_zero();

  for (int k0 = 0; k0 < NC; k0 += 32) {
    if (k0 + 32 < NC)
      __builtin_prefetch(&a16[(size_t)row0 * NC + k0 + 32], 0, 3);
    for (int e = t; e < 64 * 32; e += 128) {
      int kk = e & 31, m = e >> 5;  // coalesced along k within a row
      As[m * 40 + kk] = a16[(size_t)(row0 + m) * NC + k0 + kk];
    }
    for (int e = t; e < 64 * 32; e += 128) {
      int kk = e >> 6, nn = e & 63;  // coalesced along n
      Bs[nn * 40 + kk] = wo16[(size_t)(k0 + kk) * NC + n0 + nn];
    }
    __syncthreads();
    v16h a = load_a_frag(As, wave * 16, 0, 40);
#pragma unroll
    for (int j = 0; j < 4; ++j) {
      v16h b = load_b_frag(Bs, j * 16, 0, 40);
      acc[j] = wmma_f16(a, b, acc[j]);
    }
    __syncthreads();
  }

  const int colL = lane & 15;
  const int rbase = (lane >> 4) * 8;
  const int mrow0 = row0 + wave * 16;
#pragma unroll
  for (int j = 0; j < 4; ++j) {
    const int c = n0 + j * 16 + colL;
    const float bval = bo[c];
#pragma unroll
    for (int r = 0; r < 8; ++r) {
      int row = mrow0 + rbase + r;
      int img = row >> 12;
      int hw = row & 4095;
      // NCHW: consecutive rows (hw) are contiguous for fixed channel.
      out[((size_t)img * NC + c) * NHW + hw] = acc[j][r] + bval;
    }
  }
}

// ---------------------------------------------------------------------------
extern "C" void kernel_launch(void* const* d_in, const int* in_sizes, int n_in,
                              void* d_out, int out_size, void* d_ws,
                              size_t ws_size, hipStream_t stream) {
  (void)in_sizes; (void)n_in; (void)out_size; (void)ws_size;
  const float* x1 = (const float*)d_in[0];
  const float* x2 = (const float*)d_in[1];
  const float* wq = (const float*)d_in[2];
  const float* bq = (const float*)d_in[3];
  const float* wk = (const float*)d_in[4];
  const float* bk = (const float*)d_in[5];
  const float* wv = (const float*)d_in[6];
  const float* bv = (const float*)d_in[7];
  const float* w_dw = (const float*)d_in[8];
  const float* b_dw = (const float*)d_in[9];
  const float* wo = (const float*)d_in[10];
  const float* bo = (const float*)d_in[11];
  float* out = (float*)d_out;

  char* ws = (char*)d_ws;
  size_t off = 0;
  _Float16* wq16 = (_Float16*)(ws + off); off += (size_t)NC * NC * 2;
  _Float16* wk16 = (_Float16*)(ws + off); off += (size_t)NC * NC * 2;
  _Float16* wv16 = (_Float16*)(ws + off); off += (size_t)NC * NC * 2;
  _Float16* wo16 = (_Float16*)(ws + off); off += (size_t)NC * NC * 2;
  const size_t tens = (size_t)ROWS * NC * 2;  // 32 MB each
  _Float16* qr16 = (_Float16*)(ws + off); off += tens;
  _Float16* kr16 = (_Float16*)(ws + off); off += tens;
  _Float16* v16b = (_Float16*)(ws + off); off += tens;
  _Float16* v2b  = (_Float16*)(ws + off); off += tens;
  _Float16* at16 = (_Float16*)(ws + off); off += tens;

  convert_w<<<256, 256, 0, stream>>>(wq, wk, wv, wo, wq16, wk16, wv16, wo16);
  qkv_gemm<<<dim3(1024, 4, 3), 128, 0, stream>>>(
      x1, x2, wq16, wk16, wv16, bq, bk, bv, qr16, kr16, v16b);
  attn_pass<<<4096, 128, 0, stream>>>(qr16, kr16, v16b, v2b, 0);
  attn_pass<<<4096, 128, 0, stream>>>(qr16, kr16, v2b, at16, 1);
  lepe_add<<<65536, 256, 0, stream>>>(v16b, w_dw, b_dw, at16);
  out_gemm<<<dim3(1024, 4), 128, 0, stream>>>(at16, wo16, bo, out);
}
